// BlockU_24979529794135
// MI455X (gfx1250) — compile-verified, matrix operand order checked by reading, not run on hardware
//
#include <hip/hip_runtime.h>
#include <hip/hip_bf16.h>

// ---------------------------------------------------------------------------
// Problem constants (from the reference)
// ---------------------------------------------------------------------------
#define BATCH  16
#define DIMC   128
#define HH     56
#define WW     56
#define HWC    (HH * WW)            // 3136 tokens per batch image
#define TOK    (BATCH * HWC)        // 50176 tokens total
#define EHID   256
#define RHID   32
#define D2C    256
#define MHC    512
#define BETA_GATE 0.5f

typedef __attribute__((ext_vector_type(16))) __bf16 v16bf;
typedef __attribute__((ext_vector_type(8)))  float  v8f;

// ---------------------------------------------------------------------------
// Scalar helpers
// ---------------------------------------------------------------------------
__device__ __forceinline__ float gelu_f(float x) {
  // exact gelu: 0.5*x*(1+erf(x/sqrt(2)))
  return 0.5f * x * (1.0f + erff(x * 0.7071067811865476f));
}
__device__ __forceinline__ float softplus_f(float x) {
  return (x > 20.0f) ? x : log1pf(expf(x));
}
__device__ __forceinline__ float sigmoid_f(float x) {
  return 1.0f / (1.0f + expf(-x));
}

// ---------------------------------------------------------------------------
// WMMA fragment helpers (wave32, 16x16x32 bf16 -> f32)
// A (16xK row major in LDS): lane&15 -> M row; element j of v16bf:
//   K = k0 + (j>>3)*16 + (lane>>4)*8 + (j&7)   (two contiguous 8-elem runs)
// B: pre-packed in fragment order (see pack_b_kernel) -> one 32B vector load
// C/D (16x16 f32, 8 VGPRs): row r -> M = r + (lane>>4)*8, N = lane&15
// ---------------------------------------------------------------------------
__device__ __forceinline__ v16bf frag_a(const __bf16* A, int lda, int lane, int k0) {
  const int m  = lane & 15;
  const int kh = (lane >> 4) << 3;
  v16bf a;
#pragma unroll
  for (int j = 0; j < 16; ++j) {
    const int k = k0 + ((j >> 3) << 4) + kh + (j & 7);
    a[j] = A[m * lda + k];
  }
  return a;
}

// Packed-B fragment: contiguous aligned 32-byte read per lane.
// Layout index: (((nt * KT) + kt) * 32 + lane) * 16 + j
__device__ __forceinline__ v16bf frag_b_packed(const __bf16* __restrict__ Wp,
                                               int KT, int lane, int kt, int nt) {
  const size_t off = (((size_t)(nt * KT + kt) * 32) + lane) << 4;
  return *(const v16bf*)(Wp + off);
}

__device__ __forceinline__ v8f wmma_bf16(v16bf a, v16bf b, v8f c) {
  return __builtin_amdgcn_wmma_f32_16x16x32_bf16(
      /*neg_a=*/false, a, /*neg_b=*/false, b,
      /*c_mod=*/(short)0, c, /*reuse_a=*/false, /*reuse_b=*/false);
}

__device__ __forceinline__ void stage_a(__bf16* dst, const __bf16* src,
                                        int elems, int tid, int nthr) {
  for (int i = tid; i < elems; i += nthr) dst[i] = src[i];
}

// ---------------------------------------------------------------------------
// Kernel 0a: pack one fp32 weight matrix into bf16 WMMA-fragment order.
// Source element (k, n) accessed as src[k*sk + n*sn].
// ---------------------------------------------------------------------------
__global__ void pack_b_kernel(const float* __restrict__ src, __bf16* __restrict__ dst,
                              int K, int N, int sk, int sn) {
  const int KT = K >> 5;
  const int i = blockIdx.x * 256 + threadIdx.x;
  if (i >= K * N) return;
  const int j    = i & 15;
  const int lane = (i >> 4) & 31;
  const int tile = i >> 9;            // nt * KT + kt
  const int kt   = tile % KT;
  const int nt   = tile / KT;
  const int k = (kt << 5) + ((j >> 3) << 4) + ((lane >> 4) << 3) + (j & 7);
  const int n = (nt << 4) + (lane & 15);
  dst[i] = (__bf16)src[k * sk + n * sn];
}

// Kernel 0b: zero the per-batch sigma accumulator
__global__ void zero_sacc_kernel(float* sacc) {
  if (threadIdx.x < BATCH) sacc[threadIdx.x] = 0.0f;
}

// ---------------------------------------------------------------------------
// Kernel 1: positional depthwise 3x3 conv residual, NCHW fp32 -> NHWC fp32
// ---------------------------------------------------------------------------
__global__ void pos_kernel(const float* __restrict__ x, const float* __restrict__ w,
                           const float* __restrict__ bias, float* __restrict__ xp) {
  const int idx = blockIdx.x * 256 + threadIdx.x;     // t*128 + c
  const int c = idx & (DIMC - 1);
  const int t = idx >> 7;
  const int b   = t / HWC;
  const int rem = t - b * HWC;
  const int h   = rem / WW;
  const int wc  = rem - h * WW;
  const float* xb = x + ((size_t)(b * DIMC + c)) * HWC;   // channel plane
  const float  center = xb[h * WW + wc];
  float acc = bias[c];
#pragma unroll
  for (int dy = -1; dy <= 1; ++dy) {
#pragma unroll
    for (int dx = -1; dx <= 1; ++dx) {
      const int h2 = h + dy, w2 = wc + dx;
      if (h2 >= 0 && h2 < HH && w2 >= 0 && w2 < WW)
        acc += xb[h2 * WW + w2] * w[c * 9 + (dy + 1) * 3 + (dx + 1)];
    }
  }
  xp[idx] = center + acc;
}

// ---------------------------------------------------------------------------
// Kernel 2: LayerNorm over 128 channels, one wave32 per token.
// ---------------------------------------------------------------------------
__global__ void ln_kernel(const float* __restrict__ in, const float* __restrict__ g,
                          const float* __restrict__ bta, float* __restrict__ outf,
                          __bf16* __restrict__ outb) {
  const int lane = threadIdx.x & 31;
  const int wave = threadIdx.x >> 5;
  const int t = blockIdx.x * 8 + wave;
  const float* p = in + (size_t)t * DIMC;
  float v[4];
#pragma unroll
  for (int i = 0; i < 4; ++i) v[i] = p[lane + 32 * i];
  float s = v[0] + v[1] + v[2] + v[3];
#pragma unroll
  for (int o = 16; o > 0; o >>= 1) s += __shfl_xor(s, o, 32);
  const float mean = s * (1.0f / DIMC);
  float q = 0.0f;
#pragma unroll
  for (int i = 0; i < 4; ++i) { const float d = v[i] - mean; q += d * d; }
#pragma unroll
  for (int o = 16; o > 0; o >>= 1) q += __shfl_xor(q, o, 32);
  const float inv = rsqrtf(q * (1.0f / DIMC) + 1e-6f);
#pragma unroll
  for (int i = 0; i < 4; ++i) {
    const int c = lane + 32 * i;
    const float o = (v[i] - mean) * inv * g[c] + bta[c];
    if (outf) outf[(size_t)t * DIMC + c] = o;
    outb[(size_t)t * DIMC + c] = (__bf16)o;
  }
}

// ---------------------------------------------------------------------------
// Kernel 3: evidential head. GEMM [32tok x 128] x [128 x 256] via WMMA with
// B-fragment reuse over 2 M tiles, GELU into LDS, then [256 x 2] projection
// + softplus + sigma map.
// ---------------------------------------------------------------------------
__global__ void eh_kernel(const __bf16* __restrict__ x1b, const __bf16* __restrict__ w1,
                          const float* __restrict__ b1, const float* __restrict__ w2,
                          const float* __restrict__ b2, float* __restrict__ smap,
                          float* __restrict__ sacc) {
  __shared__ __bf16 sA[32 * DIMC];
  __shared__ float  sH[32 * EHID];
  const int tid = threadIdx.x, lane = tid & 31, wave = tid >> 5;
  const int t0 = blockIdx.x * 32;
  stage_a(sA, x1b + (size_t)t0 * DIMC, 32 * DIMC, tid, 256);
  if (tid == 0) __builtin_prefetch(w1, 0, 1);
  __syncthreads();

  const int KT = DIMC / 32;   // 4
  const int mbase = (lane >> 4) << 3;
#pragma unroll
  for (int nt2 = 0; nt2 < 2; ++nt2) {
    const int nt = wave * 2 + nt2;
    v8f acc0 = {}, acc1 = {};
#pragma unroll
    for (int kt = 0; kt < KT; ++kt) {
      const v16bf bf = frag_b_packed(w1, KT, lane, kt, nt);   // shared by 2 M tiles
      const v16bf a0 = frag_a(sA, DIMC, lane, kt * 32);
      const v16bf a1 = frag_a(sA + 16 * DIMC, DIMC, lane, kt * 32);
      acc0 = wmma_bf16(a0, bf, acc0);
      acc1 = wmma_bf16(a1, bf, acc1);
    }
    const int n = nt * 16 + (lane & 15);
    const float bias = b1[n];
#pragma unroll
    for (int r = 0; r < 8; ++r) {
      sH[(mbase + r) * EHID + n]      = gelu_f(acc0[r] + bias);
      sH[(16 + mbase + r) * EHID + n] = gelu_f(acc1[r] + bias);
    }
  }
  __syncthreads();

  // stage 2: each wave reduces 4 tokens (256-dim dot with the 2 cols of w2)
#pragma unroll
  for (int s2 = 0; s2 < 4; ++s2) {
    const int ml = wave * 4 + s2;
    float a0 = 0.0f, a1 = 0.0f;
    for (int j = lane; j < EHID; j += 32) {
      const float hv = sH[ml * EHID + j];
      a0 += hv * w2[j * 2 + 0];
      a1 += hv * w2[j * 2 + 1];
    }
#pragma unroll
    for (int o = 16; o > 0; o >>= 1) {
      a0 += __shfl_xor(a0, o, 32);
      a1 += __shfl_xor(a1, o, 32);
    }
    if (lane == 0) {
      const float e0 = softplus_f(a0 + b2[0]);
      const float e1 = softplus_f(a1 + b2[1]);
      const float S  = e0 + e1 + 2.0f;
      float sm = 2.0f / S;
      sm = fminf(fmaxf(sm, 0.0f), 1.0f);
      const int t = t0 + ml;
      smap[t] = sm;
      atomicAdd(&sacc[t / HWC], sm);
    }
  }
}

// ---------------------------------------------------------------------------
// Kernel 4: router. GEMM [64tok x 128] x [128 x 32] via WMMA + GELU into LDS,
// then 32-dim dot -> sigmoid gate scaled by (1 - sigma_map).
// ---------------------------------------------------------------------------
__global__ void router_kernel(const __bf16* __restrict__ x1b, const __bf16* __restrict__ w1,
                              const float* __restrict__ b1, const float* __restrict__ w2,
                              const float* __restrict__ b2, const float* __restrict__ smap,
                              float* __restrict__ maskp) {
  __shared__ __bf16 sA[64 * DIMC];
  __shared__ float  sH[64 * RHID];
  const int tid = threadIdx.x, lane = tid & 31, wave = tid >> 5;
  const int t0 = blockIdx.x * 64;
  stage_a(sA, x1b + (size_t)t0 * DIMC, 64 * DIMC, tid, 256);
  __syncthreads();

  const int KT = DIMC / 32;   // 4
  const int mt = wave >> 1, nt = wave & 1;
  v8f acc = {};
#pragma unroll
  for (int kt = 0; kt < KT; ++kt) {
    const v16bf a  = frag_a(sA + mt * 16 * DIMC, DIMC, lane, kt * 32);
    const v16bf bf = frag_b_packed(w1, KT, lane, kt, nt);
    acc = wmma_bf16(a, bf, acc);
  }
  const int n = nt * 16 + (lane & 15);
  const float bias = b1[n];
  const int mbase = (lane >> 4) << 3;
#pragma unroll
  for (int r = 0; r < 8; ++r)
    sH[(mt * 16 + mbase + r) * RHID + n] = gelu_f(acc[r] + bias);
  __syncthreads();

  if (tid < 64) {
    float lg = b2[0];
#pragma unroll
    for (int j = 0; j < RHID; ++j) lg += sH[tid * RHID + j] * w2[j];
    const int t = t0 + tid;
    maskp[t] = sigmoid_f(lg) * (1.0f - smap[t]);
  }
}

// ---------------------------------------------------------------------------
// Kernel 5: refine expand. [32tok x 128] x [128 x 256] via WMMA (B reuse x2),
// GELU -> bf16.
// ---------------------------------------------------------------------------
__global__ void fc1_kernel(const __bf16* __restrict__ x1b, const __bf16* __restrict__ w,
                           const float* __restrict__ bias, __bf16* __restrict__ outb) {
  __shared__ __bf16 sA[32 * DIMC];
  const int tid = threadIdx.x, lane = tid & 31, wave = tid >> 5;
  const int t0 = blockIdx.x * 32;
  stage_a(sA, x1b + (size_t)t0 * DIMC, 32 * DIMC, tid, 256);
  if (tid == 0) __builtin_prefetch(w, 0, 1);
  __syncthreads();
  const int KT = DIMC / 32;   // 4
  const int mbase = (lane >> 4) << 3;
#pragma unroll
  for (int nt2 = 0; nt2 < 2; ++nt2) {
    const int nt = wave * 2 + nt2;
    v8f acc0 = {}, acc1 = {};
#pragma unroll
    for (int kt = 0; kt < KT; ++kt) {
      const v16bf bf = frag_b_packed(w, KT, lane, kt, nt);
      const v16bf a0 = frag_a(sA, DIMC, lane, kt * 32);
      const v16bf a1 = frag_a(sA + 16 * DIMC, DIMC, lane, kt * 32);
      acc0 = wmma_bf16(a0, bf, acc0);
      acc1 = wmma_bf16(a1, bf, acc1);
    }
    const int n = nt * 16 + (lane & 15);
    const float bv = bias[n];
#pragma unroll
    for (int r = 0; r < 8; ++r) {
      outb[(size_t)(t0 + mbase + r) * D2C + n]      = (__bf16)gelu_f(acc0[r] + bv);
      outb[(size_t)(t0 + 16 + mbase + r) * D2C + n] = (__bf16)gelu_f(acc1[r] + bv);
    }
  }
}

// ---------------------------------------------------------------------------
// Kernel 6: depthwise 3x3 conv over 256 channels (NHWC bf16) + GELU.
// ---------------------------------------------------------------------------
__global__ void dw2_kernel(const __bf16* __restrict__ rb, const float* __restrict__ w,
                           const float* __restrict__ bias, __bf16* __restrict__ rc) {
  const int idx = blockIdx.x * 256 + threadIdx.x;     // t*256 + c
  const int c = idx & (D2C - 1);
  const int t = idx >> 8;
  const int b   = t / HWC;
  const int rem = t - b * HWC;
  const int h   = rem / WW;
  const int wc  = rem - h * WW;
  float acc = bias[c];
#pragma unroll
  for (int dy = -1; dy <= 1; ++dy) {
#pragma unroll
    for (int dx = -1; dx <= 1; ++dx) {
      const int h2 = h + dy, w2 = wc + dx;
      if (h2 >= 0 && h2 < HH && w2 >= 0 && w2 < WW) {
        const int tn = t + dy * WW + dx;
        acc += (float)rb[(size_t)tn * D2C + c] * w[c * 9 + (dy + 1) * 3 + (dx + 1)];
      }
    }
  }
  rc[idx] = (__bf16)gelu_f(acc);
}

// ---------------------------------------------------------------------------
// Kernel 7: refine project [32tok x 256] x [256 x 128] via WMMA (B reuse x2)
// + fused delta/gate/residual -> xs (fp32)
// ---------------------------------------------------------------------------
__global__ void fc2_kernel(const __bf16* __restrict__ rc, const __bf16* __restrict__ w,
                           const float* __restrict__ bias, const float* __restrict__ x1f,
                           const float* __restrict__ xp, const float* __restrict__ maskp,
                           const float* __restrict__ sacc, const float* __restrict__ lamp,
                           float* __restrict__ xs) {
  __shared__ __bf16 sA[32 * D2C];
  const int tid = threadIdx.x, lane = tid & 31, wave = tid >> 5;
  const int t0 = blockIdx.x * 32;
  stage_a(sA, rc + (size_t)t0 * D2C, 32 * D2C, tid, 256);
  if (tid == 0) __builtin_prefetch(w, 0, 1);
  __syncthreads();

  const int  bidx = t0 / HWC;                 // 32-token tile never straddles a batch
  const float sig  = sacc[bidx] * (1.0f / HWC);
  const float gate = 1.0f - BETA_GATE * sig;
  const float lam  = lamp[0];

  const int KT = D2C / 32;    // 8
  const int nt = wave;
  v8f acc0 = {}, acc1 = {};
#pragma unroll
  for (int kt = 0; kt < KT; ++kt) {
    const v16bf bf = frag_b_packed(w, KT, lane, kt, nt);
    const v16bf a0 = frag_a(sA, D2C, lane, kt * 32);
    const v16bf a1 = frag_a(sA + 16 * D2C, D2C, lane, kt * 32);
    acc0 = wmma_bf16(a0, bf, acc0);
    acc1 = wmma_bf16(a1, bf, acc1);
  }
  const int n = nt * 16 + (lane & 15);
  const float bv = bias[n];
  const int mbase = (lane >> 4) << 3;
#pragma unroll
  for (int r = 0; r < 8; ++r) {
    {
      const int t = t0 + mbase + r;
      const float ro = acc0[r] + bv;
      const float a1v = x1f[(size_t)t * DIMC + n];
      const float d  = maskp[t] * lam * (ro - a1v) * gate;
      xs[(size_t)t * DIMC + n] = xp[(size_t)t * DIMC + n] + a1v + d;
    }
    {
      const int t = t0 + 16 + mbase + r;
      const float ro = acc1[r] + bv;
      const float a1v = x1f[(size_t)t * DIMC + n];
      const float d  = maskp[t] * lam * (ro - a1v) * gate;
      xs[(size_t)t * DIMC + n] = xp[(size_t)t * DIMC + n] + a1v + d;
    }
  }
}

// ---------------------------------------------------------------------------
// Kernel 8: MLP fc1 [32tok x 128] x [128 x 512] via WMMA (B reuse x2),
// GELU -> bf16. Each wave owns 4 N tiles x 2 M tiles (8 accumulators).
// ---------------------------------------------------------------------------
__global__ void mlp1_kernel(const __bf16* __restrict__ x2b, const __bf16* __restrict__ w,
                            const float* __restrict__ bias, __bf16* __restrict__ hb) {
  __shared__ __bf16 sA[32 * DIMC];
  const int tid = threadIdx.x, lane = tid & 31, wave = tid >> 5;
  const int t0 = blockIdx.x * 32;
  stage_a(sA, x2b + (size_t)t0 * DIMC, 32 * DIMC, tid, 256);
  if (tid == 0) __builtin_prefetch(w, 0, 1);
  __syncthreads();

  const int KT = DIMC / 32;   // 4
  v8f acc[8] = {{}, {}, {}, {}, {}, {}, {}, {}};
#pragma unroll
  for (int kt = 0; kt < KT; ++kt) {
    const v16bf a0 = frag_a(sA, DIMC, lane, kt * 32);
    const v16bf a1 = frag_a(sA + 16 * DIMC, DIMC, lane, kt * 32);
#pragma unroll
    for (int i = 0; i < 4; ++i) {
      const v16bf bf = frag_b_packed(w, KT, lane, kt, wave * 4 + i);
      acc[2 * i]     = wmma_bf16(a0, bf, acc[2 * i]);
      acc[2 * i + 1] = wmma_bf16(a1, bf, acc[2 * i + 1]);
    }
  }
  const int mbase = (lane >> 4) << 3;
#pragma unroll
  for (int i = 0; i < 4; ++i) {
    const int n = (wave * 4 + i) * 16 + (lane & 15);
    const float bv = bias[n];
#pragma unroll
    for (int r = 0; r < 8; ++r) {
      hb[(size_t)(t0 + mbase + r) * MHC + n]      = (__bf16)gelu_f(acc[2 * i][r] + bv);
      hb[(size_t)(t0 + 16 + mbase + r) * MHC + n] = (__bf16)gelu_f(acc[2 * i + 1][r] + bv);
    }
  }
}

// ---------------------------------------------------------------------------
// Kernel 9: MLP fc2 [32tok x 512] x [512 x 128] via WMMA (B reuse x2) +
// residual + NHWC -> NCHW transpose store to d_out.
// ---------------------------------------------------------------------------
__global__ void mlp2_kernel(const __bf16* __restrict__ hb, const __bf16* __restrict__ w,
                            const float* __restrict__ bias, const float* __restrict__ xs,
                            float* __restrict__ out) {
  __shared__ __bf16 sA[32 * MHC];
  const int tid = threadIdx.x, lane = tid & 31, wave = tid >> 5;
  const int t0 = blockIdx.x * 32;
  stage_a(sA, hb + (size_t)t0 * MHC, 32 * MHC, tid, 256);
  if (tid == 0) __builtin_prefetch(w, 0, 1);
  __syncthreads();

  const int KT = MHC / 32;    // 16
  const int nt = wave;
  v8f acc0 = {}, acc1 = {};
#pragma unroll
  for (int kt = 0; kt < KT; ++kt) {
    const v16bf bf = frag_b_packed(w, KT, lane, kt, nt);
    const v16bf a0 = frag_a(sA, MHC, lane, kt * 32);
    const v16bf a1 = frag_a(sA + 16 * MHC, MHC, lane, kt * 32);
    acc0 = wmma_bf16(a0, bf, acc0);
    acc1 = wmma_bf16(a1, bf, acc1);
  }
  const int n = nt * 16 + (lane & 15);
  const float bv = bias[n];
  const int mbase = (lane >> 4) << 3;
#pragma unroll
  for (int r = 0; r < 8; ++r) {
    {
      const int t = t0 + mbase + r;
      const float val = acc0[r] + bv + xs[(size_t)t * DIMC + n];
      const int b = t / HWC, rem = t - b * HWC;
      out[((size_t)(b * DIMC + n)) * HWC + rem] = val;
    }
    {
      const int t = t0 + 16 + mbase + r;
      const float val = acc1[r] + bv + xs[(size_t)t * DIMC + n];
      const int b = t / HWC, rem = t - b * HWC;
      out[((size_t)(b * DIMC + n)) * HWC + rem] = val;
    }
  }
}

// ---------------------------------------------------------------------------
// Host launcher
// ---------------------------------------------------------------------------
extern "C" void kernel_launch(void* const* d_in, const int* in_sizes, int n_in,
                              void* d_out, int out_size, void* d_ws, size_t ws_size,
                              hipStream_t stream) {
  (void)in_sizes; (void)n_in; (void)out_size; (void)ws_size;

  const float* x      = (const float*)d_in[0];
  const float* pos_w  = (const float*)d_in[1];
  const float* pos_b  = (const float*)d_in[2];
  const float* n1_w   = (const float*)d_in[3];
  const float* n1_b   = (const float*)d_in[4];
  const float* n2_w   = (const float*)d_in[5];
  const float* n2_b   = (const float*)d_in[6];
  const float* eh_w1  = (const float*)d_in[7];
  const float* eh_b1  = (const float*)d_in[8];
  const float* eh_w2  = (const float*)d_in[9];
  const float* eh_b2  = (const float*)d_in[10];
  const float* r1_w   = (const float*)d_in[11];
  const float* r1_b   = (const float*)d_in[12];
  const float* r2_w   = (const float*)d_in[13];
  const float* r2_b   = (const float*)d_in[14];
  const float* f1_w   = (const float*)d_in[15];
  const float* f1_b   = (const float*)d_in[16];
  const float* fdw_w  = (const float*)d_in[17];
  const float* fdw_b  = (const float*)d_in[18];
  const float* f3_w   = (const float*)d_in[19];
  const float* f3_b   = (const float*)d_in[20];
  const float* lam    = (const float*)d_in[21];
  const float* m1_w   = (const float*)d_in[22];
  const float* m1_b   = (const float*)d_in[23];
  const float* m2_w   = (const float*)d_in[24];
  const float* m2_b   = (const float*)d_in[25];

  char* ws = (char*)d_ws;
  size_t off = 0;
  auto alloc = [&](size_t bytes) -> void* {
    void* p = ws + off;
    off = (off + bytes + 255) & ~(size_t)255;
    return p;
  };

  float*  xp    = (float*) alloc((size_t)TOK * DIMC * 4);   // x after pos-conv, NHWC
  float*  x1f   = (float*) alloc((size_t)TOK * DIMC * 4);   // LN1 out fp32 (attn_out)
  __bf16* x1b   = (__bf16*)alloc((size_t)TOK * DIMC * 2);   // LN1 out bf16
  float*  smap  = (float*) alloc((size_t)TOK * 4);          // sigma map per token
  float*  sacc  = (float*) alloc((size_t)BATCH * 4);        // per-batch sigma accumulator
  float*  maskp = (float*) alloc((size_t)TOK * 4);          // router mask
  __bf16* rb    = (__bf16*)alloc((size_t)TOK * D2C * 2);    // gelu(f1) bf16
  __bf16* rc    = (__bf16*)alloc((size_t)TOK * D2C * 2);    // gelu(dwconv) bf16
  float*  xs    = (float*) alloc((size_t)TOK * DIMC * 4);   // x + attn + delta
  __bf16* x2b   = (__bf16*)alloc((size_t)TOK * DIMC * 2);   // LN2 out bf16
  __bf16* hb    = (__bf16*)alloc((size_t)TOK * MHC * 2);    // MLP hidden bf16
  __bf16* w_eh1 = (__bf16*)alloc(32768 * 2);                // packed fragment order
  __bf16* w_r1  = (__bf16*)alloc(4096 * 2);
  __bf16* w_f1  = (__bf16*)alloc(32768 * 2);
  __bf16* w_f3  = (__bf16*)alloc(32768 * 2);
  __bf16* w_m1  = (__bf16*)alloc(65536 * 2);
  __bf16* w_m2  = (__bf16*)alloc(65536 * 2);

  const int NT32 = TOK / 32;   // 1568 (32-token M blocks)

  // --- pack weights into WMMA fragment order (bf16) --------------------------
  // eh_w1 [128][256] (in,out):  (k,n) = src[k*256 + n]
  pack_b_kernel<<<(DIMC * EHID) / 256, 256, 0, stream>>>(eh_w1, w_eh1, DIMC, EHID, EHID, 1);
  // r1_w [32][128] (out,in):    (k,n) = src[n*128 + k]
  pack_b_kernel<<<(DIMC * RHID) / 256, 256, 0, stream>>>(r1_w, w_r1, DIMC, RHID, 1, DIMC);
  // f1_w [256][128] (out,in):   (k,n) = src[n*128 + k]
  pack_b_kernel<<<(DIMC * D2C) / 256, 256, 0, stream>>>(f1_w, w_f1, DIMC, D2C, 1, DIMC);
  // f3_w [128][256] (out,in):   (k,n) = src[n*256 + k]
  pack_b_kernel<<<(D2C * DIMC) / 256, 256, 0, stream>>>(f3_w, w_f3, D2C, DIMC, 1, D2C);
  // m1_w [128][512] (in,out):   (k,n) = src[k*512 + n]
  pack_b_kernel<<<(DIMC * MHC) / 256, 256, 0, stream>>>(m1_w, w_m1, DIMC, MHC, MHC, 1);
  // m2_w [512][128] (in,out):   (k,n) = src[k*128 + n]
  pack_b_kernel<<<(MHC * DIMC) / 256, 256, 0, stream>>>(m2_w, w_m2, MHC, DIMC, DIMC, 1);
  zero_sacc_kernel<<<1, 32, 0, stream>>>(sacc);

  pos_kernel<<<(TOK * DIMC) / 256, 256, 0, stream>>>(x, pos_w, pos_b, xp);

  ln_kernel<<<TOK / 8, 256, 0, stream>>>(xp, n1_w, n1_b, x1f, x1b);

  eh_kernel<<<NT32, 256, 0, stream>>>(x1b, w_eh1, eh_b1, eh_w2, eh_b2, smap, sacc);

  router_kernel<<<TOK / 64, 256, 0, stream>>>(x1b, w_r1, r1_b, r2_w, r2_b, smap, maskp);

  fc1_kernel<<<NT32, 256, 0, stream>>>(x1b, w_f1, f1_b, rb);

  dw2_kernel<<<(TOK * D2C) / 256, 256, 0, stream>>>(rb, fdw_w, fdw_b, rc);

  fc2_kernel<<<NT32, 256, 0, stream>>>(rc, w_f3, f3_b, x1f, xp, maskp, sacc, lam, xs);

  ln_kernel<<<TOK / 8, 256, 0, stream>>>(xs, n2_w, n2_b, nullptr, x2b);

  mlp1_kernel<<<NT32, 256, 0, stream>>>(x2b, w_m1, m1_b, hb);

  mlp2_kernel<<<NT32, 256, 0, stream>>>(hb, w_m2, m2_b, xs, (float*)d_out);
}
